// SRUpp_42477226557453
// MI455X (gfx1250) — matrix operand order repaired and checked by reading, not compiled, not run on hardware
//
#include <hip/hip_runtime.h>
#include <hip/hip_bf16.h>

typedef __bf16 bf16;
typedef __attribute__((ext_vector_type(16))) bf16  v16bf;
typedef __attribute__((ext_vector_type(8)))  bf16  v8bf;
typedef __attribute__((ext_vector_type(8)))  float v8f;
typedef unsigned int v4u  __attribute__((ext_vector_type(4)));
typedef int          v8i_t __attribute__((ext_vector_type(8)));
typedef int          v4i_t __attribute__((ext_vector_type(4)));

#define T_DIM    2048
#define B_DIM    4
#define IN_DIM   2048
#define PROJ_DIM 1024
#define OUT_DIM  2048
#define H_DIM    8
#define HD_DIM   128
#define MROWS    (T_DIM * B_DIM)   // 8192 flattened (t,b) rows

#define LDS_ROW  40                // 32 bf16 payload + 4-DWORD TDM pad -> 80B stride (16B aligned, bank-conflict-free)

#if defined(__has_builtin)
#if __has_builtin(__builtin_amdgcn_tensor_load_to_lds) && __has_builtin(__builtin_amdgcn_s_wait_tensorcnt)
#define HAVE_TDM 1
#endif
#endif
#ifndef HAVE_TDM
#define HAVE_TDM 0
#endif

// ---------- helpers ----------

__device__ __forceinline__ bf16 to_bf16(float f) {
  unsigned u = __float_as_uint(f);
  u += 0x7FFFu + ((u >> 16) & 1u);          // round-to-nearest-even
  unsigned short s = (unsigned short)(u >> 16);
  bf16 b;
  __builtin_memcpy(&b, &s, 2);
  return b;
}

// A-operand fragment (16 rows x 32 K), ISA 7.12.2 layout:
// lane L (r=L&15,g=L>>4) holds row r: elems 0..7 = K[g*8..g*8+7], elems 8..15 = K[16+g*8..]
__device__ __forceinline__ v16bf load_a_frag(const bf16* base, int ld) {
  int lane = threadIdx.x & 31;
  int r = lane & 15, g = lane >> 4;
  const bf16* p = base + (long)r * ld + g * 8;
  union { v16bf f; v8bf h[2]; } u;
  u.h[0] = *(const v8bf*)p;
  u.h[1] = *(const v8bf*)(p + 16);
  return u.f;
}

// B-operand fragment (32 K x 16 cols, stored row-major [N,K]):
// lane L holds col r: elems 0..7 = K[g*8..g*8+7], elems 8..15 = K[16+g*8..]
__device__ __forceinline__ v16bf load_b_frag(const bf16* base, int ld) {
  int lane = threadIdx.x & 31;
  int r = lane & 15, g = lane >> 4;
  const bf16* p = base + (long)r * ld + g * 8;
  union { v16bf f; v8bf h[2]; } u;
  u.h[0] = *(const v8bf*)p;
  u.h[1] = *(const v8bf*)(p + 16);
  return u.f;
}

__device__ __forceinline__ v8f wmma_bf16(v16bf a, v16bf b, v8f c) {
  return __builtin_amdgcn_wmma_f32_16x16x32_bf16(
      /*neg_a=*/false, a, /*neg_b=*/false, b,
      /*c_mod=*/(short)0, c, /*reuse_a=*/false, /*reuse_b=*/false);
}

__device__ __forceinline__ float half16_max(float x) {
#pragma unroll
  for (int off = 1; off < 16; off <<= 1) x = fmaxf(x, __shfl_xor(x, off, 32));
  return x;
}
__device__ __forceinline__ float half16_sum(float x) {
#pragma unroll
  for (int off = 1; off < 16; off <<= 1) x += __shfl_xor(x, off, 32);
  return x;
}

#if HAVE_TDM
// Issue a TDM 2D tile DMA: global (row-major, row_stride elems of bf16) -> LDS,
// tile_h rows x 32 cols, with LDS padding 4 DWORDs per 64B row -> LDS_ROW elem stride.
// This toolchain exposes the 6-arg builtin:
//   (uint32x4 g0, int32x8 g1, int32x4 g2, int32x4 g3, int32x8 pad, i32 cpol)
__device__ __forceinline__ void tdm_load_tile(const bf16* gptr, unsigned lds_off,
                                              int tensor_w, int tensor_h,
                                              int tile_h, int row_stride) {
  unsigned long long ga = (unsigned long long)(uintptr_t)gptr;
  v4u g0;
  g0[0] = 1u;                                             // count=1 (valid), user mode
  g0[1] = lds_off;                                        // lds_addr (bytes)
  g0[2] = (unsigned)(ga & 0xFFFFFFFFull);                 // global_addr[31:0]
  g0[3] = (unsigned)((ga >> 32) & 0x1FFFFFFull) | (2u << 30); // addr[56:32] | type=2
  v8i_t g1;
  g1[0] = (1 << 16)        // data_size = 1 -> 2 bytes
        | (1 << 20)        // pad_enable
        | (3 << 22)        // pad_interval: every 16 DWORDs (64B)
        | (3 << 25);       // pad_amount: 4 DWORDs (16B) -> 80B row stride
  g1[1] = (tensor_w & 0xFFFF) << 16;                      // tensor_dim0[15:0]
  g1[2] = (int)(((unsigned)tensor_w >> 16) | (((unsigned)tensor_h & 0xFFFFu) << 16));
  g1[3] = (int)(((unsigned)tensor_h >> 16) | (32u << 16)); // tile_dim0 = 32
  g1[4] = tile_h & 0xFFFF;                                // tile_dim1 (tile_dim2 = 0)
  g1[5] = row_stride;                                     // tensor_dim0_stride[31:0]
  g1[6] = 0;
  g1[7] = 0;
  v4i_t z4 = {0, 0, 0, 0};
  v8i_t z8 = {0, 0, 0, 0, 0, 0, 0, 0};
  __builtin_amdgcn_tensor_load_to_lds(g0, g1, z4, z4, z8, 0);
}
#endif

// ---------- fp32 -> bf16 convert ----------

__global__ void f2bf_kernel(const float* __restrict__ in, bf16* __restrict__ out, long n) {
  long i = (long)blockIdx.x * blockDim.x + threadIdx.x;
  long stride = (long)gridDim.x * blockDim.x;
  for (; i < n; i += stride) out[i] = to_bf16(in[i]);
}

// ---------- generic bf16 GEMM: C[M,N] = A[M,K] * B[N,K]^T ----------
// Block: 256 threads = 8 waves (2 x 4), block tile 128 x 256, wave tile 64 x 64.
// K pipeline: TDM double-buffers A(128x32) / B(256x32) tiles into LDS; wave 0 issues
// DMA for tile k+1 and s_wait_tensorcnt's tile k; all waves compute from LDS.
// MODE 0: store fp32 C.  MODE 1: store fp32 C + bf16 Cb.
// MODE 2: kv epilogue (n<1024 -> Kout bf16 [m,1024]; else scatter into vT [b,h,HD,T]).

template <int MODE>
__global__ __launch_bounds__(256) void gemm_bf16_kernel(
    const bf16* __restrict__ A, const bf16* __restrict__ B,
    int M, int N, int K,
    float* __restrict__ C, bf16* __restrict__ Cb,
    bf16* __restrict__ Kout, bf16* __restrict__ VTout)
{
  (void)M;
  __shared__ bf16 ldsA[2][128 * LDS_ROW + 8];
  __shared__ bf16 ldsB[2][256 * LDS_ROW + 8];

  const int wave = threadIdx.x >> 5;
  const int wm = wave >> 2, wn = wave & 3;
  const long rowB0 = (long)blockIdx.x * 128;   // block row base
  const long colB0 = (long)blockIdx.y * 256;   // block col base
  const int lane = threadIdx.x & 31;
  const int nl = lane & 15, g = lane >> 4;

  v8f acc[4][4];
#pragma unroll
  for (int i = 0; i < 4; ++i)
#pragma unroll
    for (int j = 0; j < 4; ++j) acc[i][j] = (v8f)(0.0f);

#if HAVE_TDM
  if (threadIdx.x < 32) {   // wave 0 drives the DMA pipeline
    tdm_load_tile(A + rowB0 * K, (unsigned)(uintptr_t)&ldsA[0][0], K, M, 128, K);
    tdm_load_tile(B + colB0 * K, (unsigned)(uintptr_t)&ldsB[0][0], K, N, 256, K);
  }
#endif

  for (int k0 = 0; k0 < K; k0 += 32) {
    const int cur = (k0 >> 5) & 1;
#if HAVE_TDM
    if (threadIdx.x < 32) {
      if (k0 + 32 < K) {
        tdm_load_tile(A + rowB0 * K + k0 + 32, (unsigned)(uintptr_t)&ldsA[cur ^ 1][0], K, M, 128, K);
        tdm_load_tile(B + colB0 * K + k0 + 32, (unsigned)(uintptr_t)&ldsB[cur ^ 1][0], K, N, 256, K);
        __builtin_amdgcn_s_wait_tensorcnt(2);   // current batch complete, next in flight
      } else {
        __builtin_amdgcn_s_wait_tensorcnt(0);
      }
    }
    __syncthreads();                            // publish LDS tile to all waves
#else
    // Fallback: cooperative vector copy into the same padded LDS layout.
    for (int e = threadIdx.x; e < 128 * 2; e += 256) {
      int r = e >> 1, hh = e & 1;
      *(v8bf*)&ldsA[cur][r * LDS_ROW + hh * 16] =
          *(const v8bf*)(A + (rowB0 + r) * (long)K + k0 + hh * 16);
    }
    for (int e = threadIdx.x; e < 256 * 2; e += 256) {
      int r = e >> 1, hh = e & 1;
      *(v8bf*)&ldsB[cur][r * LDS_ROW + hh * 16] =
          *(const v8bf*)(B + (colB0 + r) * (long)K + k0 + hh * 16);
    }
    __syncthreads();
#endif

    v16bf af[4], bfr[4];
#pragma unroll
    for (int i = 0; i < 4; ++i)
      af[i] = load_a_frag(&ldsA[cur][(wm * 64 + 16 * i) * LDS_ROW], LDS_ROW);
#pragma unroll
    for (int j = 0; j < 4; ++j)
      bfr[j] = load_b_frag(&ldsB[cur][(wn * 64 + 16 * j) * LDS_ROW], LDS_ROW);
#pragma unroll
    for (int i = 0; i < 4; ++i)
#pragma unroll
      for (int j = 0; j < 4; ++j)
        acc[i][j] = wmma_bf16(af[i], bfr[j], acc[i][j]);
    __syncthreads();                            // buffer consumed before DMA reuses it
  }

#pragma unroll
  for (int i = 0; i < 4; ++i) {
    long m0 = rowB0 + wm * 64 + 16 * i;
#pragma unroll
    for (int j = 0; j < 4; ++j) {
      long n0 = colB0 + wn * 64 + 16 * j;
#pragma unroll
      for (int v = 0; v < 8; ++v) {
        long m = m0 + v + 8 * g;     // C layout: (m = v + 8g, n = lane&15)
        long n = n0 + nl;
        float val = acc[i][j][v];
        if (MODE == 0) {
          C[m * N + n] = val;
        } else if (MODE == 1) {
          C[m * N + n] = val;
          Cb[m * N + n] = to_bf16(val);
        } else {
          if (n < PROJ_DIM) {
            Kout[m * PROJ_DIM + n] = to_bf16(val);
          } else {
            int d = (int)n - PROJ_DIM;
            int h = d >> 7, dd = d & 127;
            int t = (int)(m >> 2), b = (int)m & 3;
            VTout[((long)((b * H_DIM + h) * HD_DIM + dd)) * T_DIM + t] = to_bf16(val);
          }
        }
      }
    }
  }
}

// ---------- flash attention (causal), one wave per 32 query rows ----------
// zb / kbuf: bf16 [MROWS, PROJ]  (row stride 4096 between consecutive t for fixed b)
// vT: bf16 [B, H, HD, T].  Epilogue: out = O/l * alpha + residual z32 (fp32).

__global__ __launch_bounds__(32) void attn_kernel(
    const bf16* __restrict__ zb, const bf16* __restrict__ kbuf,
    const bf16* __restrict__ vT, const float* __restrict__ z32,
    const float* __restrict__ alpha, float* __restrict__ attn_out)
{
  __shared__ bf16 lds_p[32 * 72];           // 32 q-rows x 64 keys, padded stride 72
  const int qb = blockIdx.x;                // rows [qb*32, qb*32+32)
  const int b  = blockIdx.y >> 3;
  const int h  = blockIdx.y & 7;
  const int lane = threadIdx.x & 31;
  const int nl = lane & 15, g = lane >> 4;
  const int ldq = B_DIM * PROJ_DIM;         // 4096

  const bf16* qbase = zb   + b * PROJ_DIM + h * HD_DIM;
  const bf16* kbase = kbuf + b * PROJ_DIM + h * HD_DIM;
  const bf16* vbase = vT + (long)((b * H_DIM + h) * HD_DIM) * T_DIM;

  v16bf qf[2][4];
#pragma unroll
  for (int i = 0; i < 2; ++i)
#pragma unroll
    for (int kc = 0; kc < 4; ++kc)
      qf[i][kc] = load_a_frag(qbase + (long)(qb * 32 + 16 * i) * ldq + kc * 32, ldq);

  v8f of[2][8];
  v8f rmax[2], rsum[2];
#pragma unroll
  for (int i = 0; i < 2; ++i) {
#pragma unroll
    for (int j = 0; j < 8; ++j) of[i][j] = (v8f)(0.0f);
    rmax[i] = (v8f)(-__builtin_inff());
    rsum[i] = (v8f)(0.0f);
  }

  const float scl = 0.08838834764831845f;   // HD^-0.5
  const int kmax = (qb * 32 + 31) >> 6;     // last key block touching the diagonal

  for (int kb = 0; kb <= kmax; ++kb) {
    // ---- S = Q * K^T (32 x 64 tile) ----
    v8f S[2][4];
#pragma unroll
    for (int i = 0; i < 2; ++i)
#pragma unroll
      for (int j = 0; j < 4; ++j) S[i][j] = (v8f)(0.0f);
#pragma unroll
    for (int j = 0; j < 4; ++j) {
      v16bf kf[4];
#pragma unroll
      for (int kc = 0; kc < 4; ++kc)
        kf[kc] = load_b_frag(kbase + (long)(kb * 64 + 16 * j) * ldq + kc * 32, ldq);
#pragma unroll
      for (int i = 0; i < 2; ++i)
#pragma unroll
        for (int kc = 0; kc < 4; ++kc)
          S[i][j] = wmma_bf16(qf[i][kc], kf[kc], S[i][j]);
    }
    // ---- scale + causal mask ----
#pragma unroll
    for (int i = 0; i < 2; ++i)
#pragma unroll
      for (int j = 0; j < 4; ++j)
#pragma unroll
        for (int v = 0; v < 8; ++v) {
          int m = qb * 32 + i * 16 + v + 8 * g;
          int n = kb * 64 + j * 16 + nl;
          float s = S[i][j][v] * scl;
          S[i][j][v] = (n <= m) ? s : -__builtin_inff();
        }
    // ---- online softmax update ----
#pragma unroll
    for (int i = 0; i < 2; ++i) {
      v8f mj, corr, ps;
#pragma unroll
      for (int v = 0; v < 8; ++v) {
        float mv = S[i][0][v];
#pragma unroll
        for (int j = 1; j < 4; ++j) mv = fmaxf(mv, S[i][j][v]);
        mj[v] = half16_max(mv);
      }
#pragma unroll
      for (int v = 0; v < 8; ++v) {
        float nm = fmaxf(rmax[i][v], mj[v]);
        corr[v] = __expf(rmax[i][v] - nm);
        rmax[i][v] = nm;
        ps[v] = 0.0f;
      }
#pragma unroll
      for (int j = 0; j < 4; ++j)
#pragma unroll
        for (int v = 0; v < 8; ++v) {
          float p = __expf(S[i][j][v] - rmax[i][v]);
          S[i][j][v] = p;
          ps[v] += p;
        }
#pragma unroll
      for (int v = 0; v < 8; ++v) {
        rsum[i][v] = rsum[i][v] * corr[v] + half16_sum(ps[v]);
      }
#pragma unroll
      for (int j8 = 0; j8 < 8; ++j8)
#pragma unroll
        for (int v = 0; v < 8; ++v) of[i][j8][v] *= corr[v];
    }
    // ---- C-layout P -> LDS (bf16) -> A-layout fragments ----
#pragma unroll
    for (int i = 0; i < 2; ++i)
#pragma unroll
      for (int j = 0; j < 4; ++j)
#pragma unroll
        for (int v = 0; v < 8; ++v)
          lds_p[(i * 16 + v + 8 * g) * 72 + j * 16 + nl] = to_bf16(S[i][j][v]);
    __syncthreads();
    v16bf pf[2][2];
#pragma unroll
    for (int i = 0; i < 2; ++i)
#pragma unroll
      for (int kc = 0; kc < 2; ++kc)
        pf[i][kc] = load_a_frag(&lds_p[(i * 16) * 72 + kc * 32], 72);
    // ---- O += P * V ----
#pragma unroll
    for (int j8 = 0; j8 < 8; ++j8) {
      v16bf vf[2];
#pragma unroll
      for (int kc = 0; kc < 2; ++kc)
        vf[kc] = load_b_frag(vbase + (long)(j8 * 16) * T_DIM + kb * 64 + kc * 32, T_DIM);
#pragma unroll
      for (int i = 0; i < 2; ++i)
#pragma unroll
        for (int kc = 0; kc < 2; ++kc)
          of[i][j8] = wmma_bf16(pf[i][kc], vf[kc], of[i][j8]);
    }
    __syncthreads();
  }

  // ---- normalize + rezero residual ----
  const float al = alpha[0];
#pragma unroll
  for (int i = 0; i < 2; ++i)
#pragma unroll
    for (int v = 0; v < 8; ++v) {
      float inv = 1.0f / rsum[i][v];
      int t = qb * 32 + i * 16 + v + 8 * g;
#pragma unroll
      for (int j8 = 0; j8 < 8; ++j8) {
        int d = j8 * 16 + nl;
        long idx = ((long)t * B_DIM + b) * PROJ_DIM + h * HD_DIM + d;
        attn_out[idx] = of[i][j8][v] * inv * al + z32[idx];
      }
    }
}

// ---------- LayerNorm over last dim (1024), one block per row ----------

__global__ __launch_bounds__(256) void ln_kernel(
    const float* __restrict__ in, const float* __restrict__ gam,
    const float* __restrict__ bet, bf16* __restrict__ out)
{
  __shared__ float sm[16];
  const long row = blockIdx.x;
  const float* p = in + row * PROJ_DIM;
  float s = 0.f, s2 = 0.f;
  for (int c = threadIdx.x; c < PROJ_DIM; c += 256) {
    float x = p[c];
    s += x; s2 += x * x;
  }
#pragma unroll
  for (int off = 1; off < 32; off <<= 1) {
    s  += __shfl_xor(s,  off, 32);
    s2 += __shfl_xor(s2, off, 32);
  }
  int wid = threadIdx.x >> 5;
  if ((threadIdx.x & 31) == 0) { sm[wid] = s; sm[8 + wid] = s2; }
  __syncthreads();
  if (threadIdx.x < 32) {
    float a  = (threadIdx.x < 8) ? sm[threadIdx.x] : 0.f;
    float b2 = (threadIdx.x < 8) ? sm[8 + threadIdx.x] : 0.f;
#pragma unroll
    for (int off = 1; off < 8; off <<= 1) {
      a  += __shfl_xor(a,  off, 32);
      b2 += __shfl_xor(b2, off, 32);
    }
    if (threadIdx.x == 0) { sm[0] = a; sm[8] = b2; }
  }
  __syncthreads();
  float mean = sm[0] * (1.0f / PROJ_DIM);
  float var  = sm[8] * (1.0f / PROJ_DIM) - mean * mean;
  float rstd = rsqrtf(var + 1e-5f);
  for (int c = threadIdx.x; c < PROJ_DIM; c += 256)
    out[row * PROJ_DIM + c] = to_bf16((p[c] - mean) * rstd * gam[c] + bet[c]);
}

// ---------- launch ----------

extern "C" void kernel_launch(void* const* d_in, const int* in_sizes, int n_in,
                              void* d_out, int out_size, void* d_ws, size_t ws_size,
                              hipStream_t stream) {
  (void)in_sizes; (void)n_in; (void)out_size; (void)ws_size;
  const float* x     = (const float*)d_in[0];
  const float* W1    = (const float*)d_in[1];
  const float* W2    = (const float*)d_in[2];
  const float* W3    = (const float*)d_in[3];
  const float* alpha = (const float*)d_in[4];
  const float* ln_g  = (const float*)d_in[5];
  const float* ln_b  = (const float*)d_in[6];
  float* out = (float*)d_out;

  char* ws = (char*)d_ws;
  size_t off = 0;
  auto alloc = [&](size_t bytes) -> void* {
    void* p = ws + off;
    off = (off + bytes + 255) & ~(size_t)255;
    return p;
  };
  bf16*  xb   = (bf16*) alloc((size_t)MROWS * IN_DIM * 2);
  bf16*  w1b  = (bf16*) alloc((size_t)PROJ_DIM * IN_DIM * 2);
  bf16*  w2b  = (bf16*) alloc((size_t)2 * PROJ_DIM * PROJ_DIM * 2);
  bf16*  w3b  = (bf16*) alloc((size_t)OUT_DIM * PROJ_DIM * 2);
  float* z32  = (float*)alloc((size_t)MROWS * PROJ_DIM * 4);
  bf16*  zb   = (bf16*) alloc((size_t)MROWS * PROJ_DIM * 2);
  bf16*  kbuf = (bf16*) alloc((size_t)MROWS * PROJ_DIM * 2);
  bf16*  vT   = (bf16*) alloc((size_t)MROWS * PROJ_DIM * 2);
  float* attn = (float*)alloc((size_t)MROWS * PROJ_DIM * 4);
  bf16*  lnb  = (bf16*) alloc((size_t)MROWS * PROJ_DIM * 2);

  f2bf_kernel<<<1024, 256, 0, stream>>>(x,  xb,  (long)MROWS * IN_DIM);
  f2bf_kernel<<<256,  256, 0, stream>>>(W1, w1b, (long)PROJ_DIM * IN_DIM);
  f2bf_kernel<<<256,  256, 0, stream>>>(W2, w2b, (long)2 * PROJ_DIM * PROJ_DIM);
  f2bf_kernel<<<256,  256, 0, stream>>>(W3, w3b, (long)OUT_DIM * PROJ_DIM);

  // GEMM1: z = x @ W1^T  (M=8192, N=1024, K=2048) -> z32 + zb
  gemm_bf16_kernel<1><<<dim3(64, 4), 256, 0, stream>>>(
      xb, w1b, MROWS, PROJ_DIM, IN_DIM, z32, zb, nullptr, nullptr);
  // GEMM2: kv = z @ W2^T (M=8192, N=2048, K=1024) -> k bf16 + v^T bf16
  gemm_bf16_kernel<2><<<dim3(64, 8), 256, 0, stream>>>(
      zb, w2b, MROWS, 2 * PROJ_DIM, PROJ_DIM, nullptr, nullptr, kbuf, vT);
  // causal flash attention + rezero residual
  attn_kernel<<<dim3(T_DIM / 32, B_DIM * H_DIM), 32, 0, stream>>>(
      zb, kbuf, vT, z32, alpha, attn);
  // post layer norm
  ln_kernel<<<MROWS, 256, 0, stream>>>(attn, ln_g, ln_b, lnb);
  // GEMM3: out = ln @ W3^T (M=8192, N=2048, K=1024)
  gemm_bf16_kernel<0><<<dim3(64, 8), 256, 0, stream>>>(
      lnb, w3b, MROWS, OUT_DIM, PROJ_DIM, out, nullptr, nullptr, nullptr);
}
